// GGNN_4063039062234
// MI455X (gfx1250) — matrix operand order repaired and checked by reading, not compiled, not run on hardware
//
#include <hip/hip_runtime.h>
#include <hip/hip_bf16.h>

typedef __attribute__((ext_vector_type(16))) __bf16 v16bf;
typedef __attribute__((ext_vector_type(8)))  float  v8f;

#define NSTEPS 5
#define Bsz 8
#define Nn 1024
#define Hh 64
#define Ee 2
#define Ktot 2048   /* N*E */
#define KT 64
#define NCHUNK (Ktot / KT)

// ---------------- adj f32 -> bf16 (once) ----------------
__global__ __launch_bounds__(256) void k_cvt_bf16(const float* __restrict__ src,
                                                  __hip_bfloat16* __restrict__ dst) {
    int i = (blockIdx.x * 256 + threadIdx.x) * 4;
    float4 v = *(const float4*)(src + i);
    union { __hip_bfloat16 q[4]; uint2 u; } o;
    o.q[0] = __float2bfloat16(v.x);
    o.q[1] = __float2bfloat16(v.y);
    o.q[2] = __float2bfloat16(v.z);
    o.q[3] = __float2bfloat16(v.w);
    *(uint2*)(dst + i) = o.u;
}

// ---------------- edge projection: edgeT[b][h][e*N+n] = (h @ W_edge + b_edge) permuted, bf16 ----------------
__global__ __launch_bounds__(256) void k_edge(const float* __restrict__ hst,
                                              const float* __restrict__ We,
                                              const float* __restrict__ be,
                                              __hip_bfloat16* __restrict__ edgeT) {
    int idx = blockIdx.x * 256 + threadIdx.x;      // b, h', e, n (n fastest -> coalesced store)
    int n  = idx & (Nn - 1);
    int e  = (idx >> 10) & (Ee - 1);
    int hc = (idx >> 11) & (Hh - 1);
    int b  = idx >> 17;
    int col = hc * Ee + e;                         // column of the (H*E) GEMM output
    const float* hrow = hst + (size_t)(b * Nn + n) * Hh;
    float acc = be[col];
#pragma unroll 8
    for (int k = 0; k < Hh; ++k) acc += hrow[k] * We[k * (Hh * Ee) + col];
    edgeT[(size_t)(b * Hh + hc) * Ktot + e * Nn + n] = __float2bfloat16(acc);
}

// ---------------- main WMMA GEMM with async global->LDS double buffering ----------------
// a_{in,out}[b] (1024x64) = adj_half[b] (1024x2048) @ edge[b] (2048x64)
__device__ __forceinline__ void async_cp16(__hip_bfloat16* lds, const __hip_bfloat16* g) {
    asm volatile("global_load_async_to_lds_b128 %0, %1, off"
                 :: "v"((__attribute__((address_space(3))) char*)lds), "v"(g)
                 : "memory");
}
__device__ __forceinline__ void async_wait0() {
    asm volatile("s_wait_asynccnt 0x0" ::: "memory");
}

__global__ __launch_bounds__(256) void k_adj_gemm(const __hip_bfloat16* __restrict__ adjb,
                                                  const __hip_bfloat16* __restrict__ edgeT,
                                                  float* __restrict__ ain,
                                                  float* __restrict__ aout) {
    const int m0 = blockIdx.x * 64;
    const int b  = blockIdx.y;
    const int s  = blockIdx.z;                     // 0 = A_in, 1 = A_out
    float* __restrict__ outp = s ? aout : ain;

    // double-buffered tiles: 64 rows x 64 K (+8 pad), row stride 144B (16B aligned)
    __shared__ __align__(16) __hip_bfloat16 As[2][64][72];
    __shared__ __align__(16) __hip_bfloat16 Bs[2][64][72];

    const int t    = threadIdx.x;
    const int lane = t & 31;
    const int wave = t >> 5;
    const int mi   = wave & 3;        // 16-row strip within 64-row tile
    const int ng   = wave >> 2;       // 0/1 -> columns ng*32 + {0,16}

    const int lr = t >> 2;            // load row 0..63
    const int lc = (t & 3) * 8;       // col chunk; this thread covers lc and lc+32

    const __hip_bfloat16* Ag = adjb  + (size_t)(b * Nn + m0 + lr) * (2 * Ktot) + s * Ktot + lc;
    const __hip_bfloat16* Bg = edgeT + (size_t)(b * Hh + lr) * Ktot + lc;

    // preload chunk 0 into buffer 0
    async_cp16(&As[0][lr][lc],      Ag);
    async_cp16(&As[0][lr][lc + 32], Ag + 32);
    async_cp16(&Bs[0][lr][lc],      Bg);
    async_cp16(&Bs[0][lr][lc + 32], Bg + 32);
    async_wait0();
    __syncthreads();

    // split accumulation chains (ks=0 vs ks=32) to avoid WMMA->WMMA D->C hazard NOPs
    v8f acc0a = {}, acc0b = {};
    v8f acc1a = {}, acc1b = {};

    const int mrow  = mi * 16 + (lane & 15);
    const int khalf = (lane >> 4) * 16;
    const int nc0   = ng * 32 + (lane & 15);
    const int kb    = (lane >> 4) * 16;

    for (int c = 0; c < NCHUNK; ++c) {
        const int cur = c & 1, nxt = cur ^ 1;
        const bool more = (c + 1) < NCHUNK;
        if (more) {
            const int k = (c + 1) * KT;
            async_cp16(&As[nxt][lr][lc],      Ag + k);
            async_cp16(&As[nxt][lr][lc + 32], Ag + k + 32);
            async_cp16(&Bs[nxt][lr][lc],      Bg + k);
            async_cp16(&Bs[nxt][lr][lc + 32], Bg + k + 32);
        }

        // issue all fragment loads first, then 4 independent-chain WMMAs
        union { v16bf v; uint4 u[2]; } fa0, fa1, fb00, fb10, fb01, fb11;
        // ks = 0 fragments
        fa0.u[0]  = *(const uint4*)&As[cur][mrow][khalf];
        fa0.u[1]  = *(const uint4*)&As[cur][mrow][8 + khalf];
        fb00.u[0] = *(const uint4*)&Bs[cur][nc0][kb];
        fb00.u[1] = *(const uint4*)&Bs[cur][nc0][kb + 8];
        fb10.u[0] = *(const uint4*)&Bs[cur][nc0 + 16][kb];
        fb10.u[1] = *(const uint4*)&Bs[cur][nc0 + 16][kb + 8];
        // ks = 32 fragments
        fa1.u[0]  = *(const uint4*)&As[cur][mrow][32 + khalf];
        fa1.u[1]  = *(const uint4*)&As[cur][mrow][40 + khalf];
        fb01.u[0] = *(const uint4*)&Bs[cur][nc0][32 + kb];
        fb01.u[1] = *(const uint4*)&Bs[cur][nc0][32 + kb + 8];
        fb11.u[0] = *(const uint4*)&Bs[cur][nc0 + 16][32 + kb];
        fb11.u[1] = *(const uint4*)&Bs[cur][nc0 + 16][32 + kb + 8];

        acc0a = __builtin_amdgcn_wmma_f32_16x16x32_bf16(false, fa0.v, false, fb00.v,
                                                        (short)0, acc0a, false, false);
        acc1a = __builtin_amdgcn_wmma_f32_16x16x32_bf16(false, fa0.v, false, fb10.v,
                                                        (short)0, acc1a, false, false);
        acc0b = __builtin_amdgcn_wmma_f32_16x16x32_bf16(false, fa1.v, false, fb01.v,
                                                        (short)0, acc0b, false, false);
        acc1b = __builtin_amdgcn_wmma_f32_16x16x32_bf16(false, fa1.v, false, fb11.v,
                                                        (short)0, acc1b, false, false);

        if (more) async_wait0();   // drain this wave's async writes before barrier signal
        __syncthreads();
    }

    // C/D layout: element r <-> row (mi*16 + r + 8*(lane/16)), col lane%16
    const int rbase = m0 + mi * 16 + (lane >> 4) * 8;
    const int c0    = ng * 32 + (lane & 15);
    float* orow = outp + (size_t)b * Nn * Hh;
#pragma unroll
    for (int r = 0; r < 8; ++r) {
        orow[(size_t)(rbase + r) * Hh + c0]      = acc0a[r] + acc0b[r];
        orow[(size_t)(rbase + r) * Hh + c0 + 16] = acc1a[r] + acc1b[r];
    }
}

// ---------------- GRU-style gate + state update (in place on h) ----------------
__global__ __launch_bounds__(256) void k_gate(const float* __restrict__ ain,
                                              const float* __restrict__ aout,
                                              float* __restrict__ hst,
                                              const float* __restrict__ Wg,
                                              const float* __restrict__ bg) {
    __shared__ float sa[4][64], so[4][64], sh[4][64], sz[4][64];
    const int rl = threadIdx.x >> 6;     // row within block
    const int c  = threadIdx.x & 63;     // output h index
    const size_t row = (size_t)blockIdx.x * 4 + rl;
    const size_t off = row * 64 + c;
    sa[rl][c] = ain[off];
    so[rl][c] = aout[off];
    sh[rl][c] = hst[off];
    __syncthreads();

    float s12 = bg[c];
#pragma unroll 4
    for (int k = 0; k < 64; ++k) s12 += sa[rl][k] * Wg[k * 64 + c];
#pragma unroll 4
    for (int k = 0; k < 64; ++k) s12 += so[rl][k] * Wg[(64 + k) * 64 + c];
    float g = s12;
#pragma unroll 4
    for (int k = 0; k < 64; ++k) g += sh[rl][k] * Wg[(128 + k) * 64 + c];
    float z = 1.0f / (1.0f + __expf(-g));
    sz[rl][c] = z;
    __syncthreads();

    float g2 = s12;   // first 128 rows of W_gate see identical a_in/a_out -> reuse partial sum
#pragma unroll 4
    for (int k = 0; k < 64; ++k) g2 += sz[rl][k] * sh[rl][k] * Wg[(128 + k) * 64 + c];
    float ht = tanhf(g2);
    hst[off] = (1.0f - z) * sh[rl][c] + z * ht;
}

// ---------------- output head ----------------
__global__ __launch_bounds__(256) void k_final(const float* __restrict__ hst,
                                               const float* __restrict__ ann,
                                               const float* __restrict__ Wh,
                                               const float* __restrict__ bh,
                                               const float* __restrict__ Wo,
                                               const float* __restrict__ bo,
                                               float* __restrict__ out) {
    __shared__ float sha[4][72];
    __shared__ float red[4][64];
    const int rl = threadIdx.x >> 6;
    const int c  = threadIdx.x & 63;
    const size_t row = (size_t)blockIdx.x * 4 + rl;
    sha[rl][c] = hst[row * 64 + c];
    if (c < 8) sha[rl][64 + c] = ann[row * 8 + c];
    __syncthreads();

    float g = bh[c];
#pragma unroll 4
    for (int k = 0; k < 72; ++k) g += sha[rl][k] * Wh[k * 64 + c];
    red[rl][c] = tanhf(g) * Wo[c];
    __syncthreads();
    for (int st = 32; st > 0; st >>= 1) {
        if (c < st) red[rl][c] += red[rl][c + st];
        __syncthreads();
    }
    if (c == 0) out[row] = red[rl][0] + bo[0];
}

extern "C" void kernel_launch(void* const* d_in, const int* in_sizes, int n_in,
                              void* d_out, int out_size, void* d_ws, size_t ws_size,
                              hipStream_t stream) {
    const float* initial_state = (const float*)d_in[0];
    const float* annotation    = (const float*)d_in[1];
    const float* adj           = (const float*)d_in[2];
    const float* W_edge        = (const float*)d_in[3];
    const float* b_edge        = (const float*)d_in[4];
    const float* W_gate        = (const float*)d_in[5];
    const float* b_gate        = (const float*)d_in[6];
    const float* W_hid         = (const float*)d_in[7];
    const float* b_hid         = (const float*)d_in[8];
    const float* W_out         = (const float*)d_in[9];
    const float* b_out         = (const float*)d_in[10];
    float* out = (float*)d_out;

    char* ws = (char*)d_ws;
    __hip_bfloat16* adjb  = (__hip_bfloat16*)ws;                       // 8*1024*4096*2 = 64 MiB
    __hip_bfloat16* edgeT = (__hip_bfloat16*)(ws + 67108864);          // 8*64*2048*2  =  2 MiB
    float* ain  = (float*)(ws + 69206016);                             // 2 MiB
    float* aout = (float*)(ws + 71303168);                             // 2 MiB
    float* hbuf = (float*)(ws + 73400320);                             // 2 MiB

    const int adj_elems = Bsz * Nn * 2 * Ktot;                         // 33,554,432
    k_cvt_bf16<<<adj_elems / (256 * 4), 256, 0, stream>>>(adj, adjb);
    hipMemcpyAsync(hbuf, initial_state, sizeof(float) * Bsz * Nn * Hh,
                   hipMemcpyDeviceToDevice, stream);

    for (int step = 0; step < NSTEPS; ++step) {
        k_edge<<<(Bsz * Nn * Hh * Ee) / 256, 256, 0, stream>>>(hbuf, W_edge, b_edge, edgeT);
        k_adj_gemm<<<dim3(Nn / 64, Bsz, 2), 256, 0, stream>>>(adjb, edgeT, ain, aout);
        k_gate<<<(Bsz * Nn) / 4, 256, 0, stream>>>(ain, aout, hbuf, W_gate, b_gate);
    }
    k_final<<<(Bsz * Nn) / 4, 256, 0, stream>>>(hbuf, annotation, W_hid, b_hid, W_out, b_out, out);
}
